// SAB_3015067042143
// MI455X (gfx1250) — matrix operand order repaired and checked by reading, not compiled, not run on hardware
//
#include <hip/hip_runtime.h>
#include <hip/hip_bf16.h>
#include <math.h>

typedef __attribute__((ext_vector_type(16))) _Float16 v16h;
typedef __attribute__((ext_vector_type(8)))  _Float16 v8h;
typedef __attribute__((ext_vector_type(8)))  float    v8f;
typedef __attribute__((ext_vector_type(4)))  float    v4f;

#define BB   8
#define CC   256
#define HH   128
#define WW   128
#define RR   512
#define GG   16
#define FPG  16
#define HWPX (HH*WW)          // 16384
#define PH   (HH+2)           // 130
#define EPSV 1e-5f

// ---------------- workspace layout (floats unless noted) ----------------
// S      : BB*GG*PH*PH            = 2,163,200
// K9     : BB*CC*9                = 18,432
// Gm,Bt,At,Sc,Sh : BB*CC each     = 2,048 each
// W2h    : CC*CC _Float16         = 65,536 halfs
struct WS {
    float*     S;
    float*     K9;
    float*     Gm;
    float*     Bt;
    float*     At;
    float*     Sc;   // rstd*gamma          (folded IN scale)
    float*     Sh;   // beta - mean*rstd*g  (folded IN shift)
    _Float16*  W2h;
};
__host__ __device__ inline WS ws_layout(void* base) {
    WS w;
    float* f = (float*)base;
    w.S  = f;                 f += (size_t)BB*GG*PH*PH;
    w.K9 = f;                 f += (size_t)BB*CC*9;
    w.Gm = f;                 f += BB*CC;
    w.Bt = f;                 f += BB*CC;
    w.At = f;                 f += BB*CC;
    w.Sc = f;                 f += BB*CC;
    w.Sh = f;                 f += BB*CC;
    w.W2h = (_Float16*)f;
    return w;
}

__device__ inline float leaky01(float v) { return v > 0.0f ? v : 0.1f * v; }

// ============ Kernel 1: dynamic params K9 / gamma / beta ============
__global__ void k_params(const float* __restrict__ rep,
                         const float* __restrict__ w_key,
                         const float* __restrict__ w_gamma,
                         const float* __restrict__ w_beta,
                         float* __restrict__ K9,
                         float* __restrict__ Gm,
                         float* __restrict__ Bt) {
    int t = blockIdx.x * blockDim.x + threadIdx.x;
    const int per_b = CC*9 + CC + CC;          // 2816
    if (t >= BB * per_b) return;
    int b = t / per_b, j = t % per_b;
    const v4f* r = (const v4f*)(rep + (size_t)b * RR);
    const float* wrow; float* dst; int o;
    if (j < CC*9)            { wrow = w_key   + (size_t)j*RR;        dst = K9 + (size_t)b*CC*9 + j; }
    else if (j < CC*9 + CC)  { o = j - CC*9;  wrow = w_gamma + (size_t)o*RR; dst = Gm + b*CC + o; }
    else                     { o = j - CC*9 - CC; wrow = w_beta + (size_t)o*RR; dst = Bt + b*CC + o; }
    const v4f* w4 = (const v4f*)wrow;
    v4f acc4 = {};
    #pragma unroll 4
    for (int k = 0; k < RR/4; ++k) {
        v4f a = r[k], w = w4[k];
        acc4 += a * w;
    }
    float acc = acc4.x + acc4.y + acc4.z + acc4.w;
    *dst = leaky01(acc);
}

// ============ Kernel 2: channel attention (PReLU -> sigmoid) ============
__global__ void k_att(const float* __restrict__ rep,
                      const float* __restrict__ w_ca1,
                      const float* __restrict__ prelu_a,
                      const float* __restrict__ w_ca2,
                      float* __restrict__ At) {
    __shared__ float hmid[RR/4];               // 128
    int b = blockIdx.x, tid = threadIdx.x;
    const float* r = rep + (size_t)b * RR;
    float a = prelu_a[0];
    {
        const float* wrow = w_ca1 + (size_t)tid * RR;
        float acc = 0.0f;
        for (int k = 0; k < RR; ++k) acc = fmaf(r[k], wrow[k], acc);
        hmid[tid] = acc > 0.0f ? acc : a * acc;
    }
    __syncthreads();
    for (int o = tid; o < CC; o += RR/4) {
        const float* wrow = w_ca2 + (size_t)o * (RR/4);
        float acc = 0.0f;
        for (int k = 0; k < RR/4; ++k) acc = fmaf(hmid[k], wrow[k], acc);
        At[b*CC + o] = 1.0f / (1.0f + __expf(-acc));
    }
}

// ============ Kernel 3: W2 -> f16 ============
__global__ void k_w2h(const float* __restrict__ w2, _Float16* __restrict__ w2h) {
    int t = blockIdx.x * blockDim.x + threadIdx.x;
    if (t < CC*CC) w2h[t] = (_Float16)w2[t];
}

// ============ Kernel 4: group pre-sum with reflect halo ============
__global__ void k_gsum(const float* __restrict__ x, float* __restrict__ S) {
    int blk = blockIdx.x;
    int b = blk / GG, g = blk % GG;
    const float* xb = x + ((size_t)b*CC + g*FPG) * HWPX;
    float* sp = S + (size_t)blk * PH * PH;
    for (int p = threadIdx.x; p < PH*PH; p += blockDim.x) {
        int yy = p / PH, xx = p % PH;
        int sy = yy - 1; if (sy < 0) sy = -sy; if (sy > HH-1) sy = 2*(HH-1) - sy;
        int sx = xx - 1; if (sx < 0) sx = -sx; if (sx > WW-1) sx = 2*(WW-1) - sx;
        const float* px = xb + (size_t)sy * WW + sx;
        float acc = 0.0f;
        #pragma unroll
        for (int i = 0; i < FPG; ++i) acc += px[(size_t)i * HWPX];
        sp[p] = acc;
    }
}

// ============ Kernel 5: 9-tap conv + IN stats folded to scale/shift ============
// one block (256 thr) per (b,o) plane; pre-activation written into d_out
__global__ void k_conv_stats(const float* __restrict__ S,
                             const float* __restrict__ K9,
                             const float* __restrict__ Gm,
                             const float* __restrict__ Bt,
                             float* __restrict__ pre,
                             float* __restrict__ Sc, float* __restrict__ Sh) {
    __shared__ float rsum[256];
    __shared__ float rsq[256];
    int blk = blockIdx.x;
    int b = blk / CC, o = blk % CC, g = o >> 4;
    const float* sp = S + ((size_t)b*GG + g) * PH * PH;
    const float* tp = K9 + (size_t)b*CC*9 + o*9;
    float t0=tp[0],t1=tp[1],t2=tp[2],t3=tp[3],t4=tp[4],t5=tp[5],t6=tp[6],t7=tp[7],t8=tp[8];
    float* op = pre + (size_t)blk * HWPX;
    float lsum = 0.0f, lsq = 0.0f;
    for (int p = threadIdx.x; p < HWPX; p += 256) {
        int y = p >> 7, xcol = p & 127;
        const float* r0 = sp + (size_t)y * PH + xcol;
        const float* r1 = r0 + PH;
        const float* r2 = r1 + PH;
        float acc = t0*r0[0] + t1*r0[1] + t2*r0[2]
                  + t3*r1[0] + t4*r1[1] + t5*r1[2]
                  + t6*r2[0] + t7*r2[1] + t8*r2[2];
        op[p] = acc;
        lsum += acc; lsq = fmaf(acc, acc, lsq);
    }
    rsum[threadIdx.x] = lsum; rsq[threadIdx.x] = lsq;
    __syncthreads();
    for (int s = 128; s > 0; s >>= 1) {
        if (threadIdx.x < s) { rsum[threadIdx.x] += rsum[threadIdx.x+s]; rsq[threadIdx.x] += rsq[threadIdx.x+s]; }
        __syncthreads();
    }
    if (threadIdx.x == 0) {
        float mean = rsum[0] * (1.0f / HWPX);
        float var  = rsq[0] * (1.0f / HWPX) - mean * mean;
        float rstd = rsqrtf(var + EPSV);
        float sc = rstd * Gm[blk];
        Sc[blk] = sc;
        Sh[blk] = Bt[blk] - mean * sc;     // (pre*sc + sh) == normalized+affine
    }
}

// ============ Kernel 6: fused norm/ReLU/residual + 1x1-conv WMMA GEMM ============
// one block (256 thr = 8 waves) per 64-pixel tile; computes all 256 out chans.
// In-place on d_out: whole activation tile staged to LDS (f16) before writes.
#define NTILE 64
#define LDSTR 272   // 256 + 16 halfs pad -> 544B row, keeps 32B-aligned segments
__global__ void k_gemm(const float* __restrict__ x,
                       float* __restrict__ out,          // holds pre on entry
                       const _Float16* __restrict__ W2h,
                       const float* __restrict__ bias,
                       const float* __restrict__ At,
                       const float* __restrict__ Sc, const float* __restrict__ Sh) {
    __shared__ _Float16 act[NTILE * LDSTR];               // ~34.8 KB
    int blk = blockIdx.x;
    int b  = blk / (HWPX / NTILE);
    int p0 = (blk % (HWPX / NTILE)) * NTILE;
    int tid = threadIdx.x;

    // ---- stage activation tile: (pre*sc+sh) -> ReLU -> +x*att, cvt f16 ----
    // float4-vectorized: 16 iterations, 16 float4 (=64 px) per channel row
    {
        const v4f* pre4 = (const v4f*)out;
        const v4f* x4   = (const v4f*)x;
        #pragma unroll
        for (int it = 0; it < 16; ++it) {
            int e = it * 256 + tid;                       // 4096 float4 total
            int c = e >> 4;                               // channel 0..255
            int q = e & 15;                               // float4 index in row
            size_t g4 = (((size_t)b*CC + c) * HWPX + p0) >> 2;
            v4f pv = pre4[g4 + q];
            v4f xv = x4[g4 + q];
            int scix = b*CC + c;
            float sc = Sc[scix], sh = Sh[scix], at = At[scix];
            _Float16* dst = act + (size_t)(q*4) * LDSTR + c;
            #pragma unroll
            for (int u = 0; u < 4; ++u) {
                float v = fmaf(pv[u], sc, sh);
                v = v > 0.0f ? v : 0.0f;
                v = fmaf(xv[u], at, v);
                dst[(size_t)u * LDSTR] = (_Float16)v;
            }
        }
    }
    __syncthreads();

    // ---- WMMA: M=256 (16 mt) x N=64 (nt fixed per wave), K=256 (8x32) ----
    int wave = tid >> 5, lane = tid & 31;
    int mrow = lane & 15, hi = lane >> 4;
    int nt = wave & 3;                                    // wave-invariant N tile
    union AU { v16h v; struct { v8h lo, hi; } s; };

    // Preload all 8 B fragments (activation tile) from LDS into registers:
    // B 32x16 layout chosen here (A uses same k order): lane n holds col n; halfs = 16 contiguous K.
    v16h bfr[8];
    {
        const _Float16* brow = act + (size_t)(nt*16 + mrow) * LDSTR + hi*16;
        #pragma unroll
        for (int kt = 0; kt < 8; ++kt)
            bfr[kt] = *(const v16h*)(brow + kt*32);
    }

    // Each wave: 8 M-tiles (mt = (wave>>2) + 2*i), software-pipelined A stream.
    for (int i = 0; i < 8; ++i) {
        int mt = (wave >> 2) + 2*i;                       // 0..15 across 2 waves
        const _Float16* wrow = W2h + (size_t)(mt*16 + mrow) * CC + hi*16;
        if (i < 7)                                        // hint next M-tile row
            __builtin_prefetch(wrow + 32*CC, 0, 1);
        AU a[2];
        a[0].s.lo = *(const v8h*)(wrow);
        a[0].s.hi = *(const v8h*)(wrow + 8);
        v8f acc = {};
        #pragma unroll
        for (int kt = 0; kt < 8; ++kt) {
            int cur = kt & 1, nxt = cur ^ 1;
            if (kt < 7) {                                 // pipeline: load kt+1
                a[nxt].s.lo = *(const v8h*)(wrow + (kt+1)*32);
                a[nxt].s.hi = *(const v8h*)(wrow + (kt+1)*32 + 8);
            }
            acc = __builtin_amdgcn_wmma_f32_16x16x32_f16(
                false, a[cur].v, false, bfr[kt], (short)0, acc, false, false);
        }
        // ---- write D tile (+bias): 2 rows x 16 px per store ----
        #pragma unroll
        for (int r = 0; r < 8; ++r) {
            int mo = mt*16 + r + hi*8;
            out[((size_t)b*CC + mo) * HWPX + p0 + nt*16 + mrow] = acc[r] + bias[mo];
        }
    }
}

extern "C" void kernel_launch(void* const* d_in, const int* in_sizes, int n_in,
                              void* d_out, int out_size, void* d_ws, size_t ws_size,
                              hipStream_t stream) {
    const float* x       = (const float*)d_in[0];
    const float* rep     = (const float*)d_in[1];
    const float* w_key   = (const float*)d_in[2];
    const float* w_gamma = (const float*)d_in[3];
    const float* w_beta  = (const float*)d_in[4];
    const float* w_ca1   = (const float*)d_in[5];
    const float* prelu_a = (const float*)d_in[6];
    const float* w_ca2   = (const float*)d_in[7];
    const float* w_conv2 = (const float*)d_in[8];
    const float* b_conv2 = (const float*)d_in[9];
    float* out = (float*)d_out;
    WS w = ws_layout(d_ws);

    // 1) dynamic conv kernels + gamma/beta
    {
        int total = BB * (CC*9 + CC + CC);
        k_params<<<(total + 255)/256, 256, 0, stream>>>(rep, w_key, w_gamma, w_beta,
                                                        w.K9, w.Gm, w.Bt);
    }
    // 2) channel attention
    k_att<<<BB, RR/4, 0, stream>>>(rep, w_ca1, prelu_a, w_ca2, w.At);
    // 3) W2 -> f16
    k_w2h<<<(CC*CC + 255)/256, 256, 0, stream>>>(w_conv2, w.W2h);
    // 4) group pre-sum with reflect halo
    k_gsum<<<BB*GG, 256, 0, stream>>>(x, w.S);
    // 5) 9-tap conv + instance-norm stats -> folded scale/shift (pre into d_out)
    k_conv_stats<<<BB*CC, 256, 0, stream>>>(w.S, w.K9, w.Gm, w.Bt, out, w.Sc, w.Sh);
    // 6) fused epilogue + 1x1-conv WMMA GEMM (in-place on d_out)
    k_gemm<<<BB * (HWPX / NTILE), 256, 0, stream>>>(x, out, w.W2h, b_conv2,
                                                    w.At, w.Sc, w.Sh);
}
